// Block_52948356825793
// MI455X (gfx1250) — matrix operand order repaired and checked by reading, not compiled, not run on hardware
//
#include <hip/hip_runtime.h>
#include <hip/hip_bf16.h>

// ---------------------------------------------------------------------------
// Transformer block forward for MI455X (gfx1250), wave32 + WMMA f16->f32,
// TDM (tensor_load_to_lds) double-buffered GEMM staging when available.
// B=4 T=2048 H=1024 NH=16 HD=64 FF=4096, M = B*T = 8192.
// ---------------------------------------------------------------------------

#define BB 4
#define TT 2048
#define HH 1024
#define NHEAD 16
#define HDIM 64
#define FFDIM 4096
#define MM (BB * TT)   // 8192

typedef __attribute__((ext_vector_type(16))) _Float16 v16h;
typedef __attribute__((ext_vector_type(8)))  float    v8f;
typedef __attribute__((ext_vector_type(4)))  unsigned int u32x4;
typedef __attribute__((ext_vector_type(8)))  int      i32x8;
typedef __attribute__((ext_vector_type(4)))  int      i32x4;

#if defined(__has_builtin)
#  if __has_builtin(__builtin_amdgcn_tensor_load_to_lds)
#    define HAVE_TDM 1
#  else
#    define HAVE_TDM 0
#  endif
#else
#  define HAVE_TDM 0
#endif
#if HAVE_TDM && defined(__clang_major__) && (__clang_major__ >= 23)
#  define TDM_ARGS6 1
#else
#  define TDM_ARGS6 0
#endif

__device__ __forceinline__ void wait_tensorcnt0() {
#if defined(__has_builtin) && __has_builtin(__builtin_amdgcn_s_wait_tensorcnt)
    __builtin_amdgcn_s_wait_tensorcnt(0);
#else
    asm volatile("s_wait_tensorcnt 0x0" ::: "memory");
#endif
}

__device__ __forceinline__ v8f zero8() {
    v8f z;
#pragma unroll
    for (int i = 0; i < 8; ++i) z[i] = 0.0f;
    return z;
}

#if HAVE_TDM
// Build a 2D-tile TDM descriptor (D#) per CDNA5 ISA ch.10 §8.3/8.4 and issue it.
//   lds_off        : LDS byte offset of tile destination
//   gaddr          : global byte address of tile start
//   ld_elems       : tensor dim0 length == dim0 stride (elements, 2B each)
//   rows           : tensor dim1 length (elements)
//   tile_w, tile_h : tile dims (elements)
//   pad_int_code   : pad_interval code (3 -> every 16 DWORDs, 5 -> every 64 DWORDs)
//   pad_amt_code   : pad_amount code   (3 -> 4 DWORDs = 8 halves)
__device__ __forceinline__ void tdm_load_2d(unsigned lds_off, const void* gaddr,
                                            unsigned ld_elems, unsigned rows,
                                            unsigned tile_w, unsigned tile_h,
                                            unsigned pad_int_code, unsigned pad_amt_code) {
    const unsigned long long ga = (unsigned long long)gaddr;
    u32x4 g0;
    g0[0] = 1u;                                          // count=1, user descriptor
    g0[1] = lds_off;                                     // lds_addr (bytes)
    g0[2] = (unsigned)(ga & 0xFFFFFFFFu);                // global_addr[31:0]
    g0[3] = (unsigned)((ga >> 32) & 0x1FFFFFFu)          // global_addr[56:32]
          | (2u << 30);                                  // type=2 ("image")
    i32x8 g1;
    unsigned d0 = (1u << 16)                             // data_size=1 -> 2 bytes
                | (1u << 20)                             // pad_enable
                | (pad_int_code << 22)                   // pad_interval
                | (pad_amt_code << 25);                  // pad_amount
    g1[0] = (int)d0;                                     // workgroup_mask=0 (not in cluster)
    g1[1] = (int)((ld_elems & 0xFFFFu) << 16);           // tensor_dim0[15:0]
    g1[2] = (int)(((ld_elems >> 16) & 0xFFFFu)           // tensor_dim0[31:16]
          | ((rows & 0xFFFFu) << 16));                   // tensor_dim1[15:0]
    g1[3] = (int)(((rows >> 16) & 0xFFFFu)               // tensor_dim1[31:16]
          | (tile_w << 16));                             // tile_dim0
    g1[4] = (int)(tile_h & 0xFFFFu);                     // tile_dim1 (tile_dim2=0)
    g1[5] = (int)ld_elems;                               // tensor_dim0_stride[31:0]
    g1[6] = 0;                                           // stride hi / dim1_stride lo
    g1[7] = 0;
    const i32x4 z4 = {0, 0, 0, 0};
#if TDM_ARGS6
    const i32x8 z8 = {0, 0, 0, 0, 0, 0, 0, 0};
    __builtin_amdgcn_tensor_load_to_lds(g0, g1, z4, z4, z8, 0);
#else
    __builtin_amdgcn_tensor_load_to_lds(g0, g1, z4, z4, 0);
#endif
}
#endif  // HAVE_TDM

// A fragment (16x32 f16, row-major src, ld in halves).
__device__ __forceinline__ v16h load_frag_a(const _Float16* __restrict__ src, int ld) {
    const int lane = threadIdx.x & 31;
    const int row  = lane & 15;
    const int kh   = (lane >> 4) << 4;   // 0 or 16
    const _Float16* p = src + row * ld + kh;
    v16h a;
#pragma unroll
    for (int i = 0; i < 8; ++i) a[i] = p[i];
#pragma unroll
    for (int i = 0; i < 8; ++i) a[8 + i] = p[8 + i];
    return a;
}

// B fragment (32x16 f16, row-major src B[k][n], ld in halves).
__device__ __forceinline__ v16h load_frag_b(const _Float16* __restrict__ src, int ld) {
    const int lane = threadIdx.x & 31;
    const int col  = lane & 15;
    const int kb   = (lane >> 4) << 4;   // 0 or 16
    v16h b;
#pragma unroll
    for (int i = 0; i < 16; ++i) b[i] = src[(kb + i) * ld + col];
    return b;
}

__device__ __forceinline__ float gelu_tanh(float x) {
    const float x3 = x * x * x;
    return 0.5f * x * (1.0f + tanhf(0.7978845608028654f * (x + 0.044715f * x3)));
}

// ---------------------------------------------------------------------------
__global__ void cvt_f32_to_f16(const float* __restrict__ s, _Float16* __restrict__ d, size_t n) {
    size_t i = (size_t)blockIdx.x * blockDim.x + threadIdx.x;
    const size_t stride = (size_t)gridDim.x * blockDim.x;
    for (; i < n; i += stride) d[i] = (_Float16)s[i];
}

__global__ void add_residual(const float* __restrict__ x, float* __restrict__ attn_xr,
                             _Float16* __restrict__ xrh, size_t n) {
    size_t i = (size_t)blockIdx.x * blockDim.x + threadIdx.x;
    const size_t stride = (size_t)gridDim.x * blockDim.x;
    for (; i < n; i += stride) {
        const float v = x[i] + attn_xr[i];
        attn_xr[i] = v;
        xrh[i] = (_Float16)v;
    }
}

// ---------------------------------------------------------------------------
// WMMA GEMM: C[M,N] = A[M,K] @ W[K,N] + bias.
//   EPI==0: store f16; EPI==1: gelu, f16; EPI==2: +resid, f32 out.
// Block tile 128x128, BK=32, 256 threads = 8 waves (4x2), wave tile 32x64.
// TDM path: wave 0 DMAs the next K-tile into the ping-pong LDS buffer while
// all waves run WMMAs on the current one; fenced by s_wait_tensorcnt.
// ---------------------------------------------------------------------------
template <int EPI>
__global__ __launch_bounds__(256) void gemm_wmma(const _Float16* __restrict__ A,
                                                 const _Float16* __restrict__ W,
                                                 const float* __restrict__ bias,
                                                 void* __restrict__ Out,
                                                 const float* __restrict__ resid,
                                                 int M, int N, int K) {
    const int tid  = threadIdx.x;
    const int wave = tid >> 5;
    const int wm   = wave >> 1;                 // 0..3 -> 32-row group
    const int wn   = wave & 1;                  // 0..1 -> 64-col group
    const long rowBase = (long)blockIdx.y * 128;
    const long colBase = (long)blockIdx.x * 128;

    v8f acc[2][4];
#pragma unroll
    for (int i = 0; i < 2; ++i)
#pragma unroll
        for (int j = 0; j < 4; ++j) acc[i][j] = zero8();

#if HAVE_TDM
    __shared__ __align__(16) _Float16 Ash[2][128 * 40];
    __shared__ __align__(16) _Float16 Bsh[2][32 * 136];

    if (wave == 0) {
        tdm_load_2d((unsigned)(unsigned long long)&Ash[0][0],
                    A + (size_t)rowBase * K, K, M, 32, 128, /*pad*/ 3, 3);
        tdm_load_2d((unsigned)(unsigned long long)&Bsh[0][0],
                    W + colBase, N, K, 128, 32, /*pad*/ 5, 3);
        wait_tensorcnt0();
    }
    __syncthreads();

    int cur = 0;
    for (int k0 = 0; k0 < K; k0 += 32) {
        if (wave == 0 && k0 + 32 < K) {
            tdm_load_2d((unsigned)(unsigned long long)&Ash[cur ^ 1][0],
                        A + (size_t)rowBase * K + (k0 + 32), K, M, 32, 128, 3, 3);
            tdm_load_2d((unsigned)(unsigned long long)&Bsh[cur ^ 1][0],
                        W + (size_t)(k0 + 32) * N + colBase, N, K, 128, 32, 5, 3);
        }

        v16h af[2], bf[4];
#pragma unroll
        for (int i = 0; i < 2; ++i)
            af[i] = load_frag_a(&Ash[cur][(wm * 32 + i * 16) * 40], 40);
#pragma unroll
        for (int j = 0; j < 4; ++j)
            bf[j] = load_frag_b(&Bsh[cur][wn * 64 + j * 16], 136);

#pragma unroll
        for (int i = 0; i < 2; ++i)
#pragma unroll
            for (int j = 0; j < 4; ++j)
                acc[i][j] = __builtin_amdgcn_wmma_f32_16x16x32_f16(
                    false, af[i], false, bf[j], (short)0, acc[i][j], false, false);

        if (wave == 0 && k0 + 32 < K) wait_tensorcnt0();
        __syncthreads();
        cur ^= 1;
    }
#else
    __shared__ __align__(16) _Float16 Ash[128 * 40];
    __shared__ __align__(16) _Float16 Bsh[32 * 136];

    for (int k0 = 0; k0 < K; k0 += 32) {
        if (k0) __syncthreads();
#pragma unroll
        for (int c = tid; c < 512; c += 256) {
            const int r  = c >> 2;
            const int kc = (c & 3) << 3;
            *(uint4*)&Ash[r * 40 + kc] =
                *(const uint4*)&A[(size_t)(rowBase + r) * K + k0 + kc];
            if (k0 + 32 < K)
                __builtin_prefetch(&A[(size_t)(rowBase + r) * K + k0 + 32 + kc], 0, 0);
        }
#pragma unroll
        for (int c = tid; c < 512; c += 256) {
            const int r  = c >> 4;
            const int nc = (c & 15) << 3;
            *(uint4*)&Bsh[r * 136 + nc] =
                *(const uint4*)&W[(size_t)(k0 + r) * N + colBase + nc];
            if (k0 + 32 < K)
                __builtin_prefetch(&W[(size_t)(k0 + 32 + r) * N + colBase + nc], 0, 0);
        }
        __syncthreads();

        v16h af[2], bf[4];
#pragma unroll
        for (int i = 0; i < 2; ++i) af[i] = load_frag_a(&Ash[(wm * 32 + i * 16) * 40], 40);
#pragma unroll
        for (int j = 0; j < 4; ++j) bf[j] = load_frag_b(&Bsh[wn * 64 + j * 16], 136);

#pragma unroll
        for (int i = 0; i < 2; ++i)
#pragma unroll
            for (int j = 0; j < 4; ++j)
                acc[i][j] = __builtin_amdgcn_wmma_f32_16x16x32_f16(
                    false, af[i], false, bf[j], (short)0, acc[i][j], false, false);
    }
#endif

    // epilogue: C layout -> VGPR r holds row (r + 8*hi), lane&15 holds col
    const int lane = tid & 31;
    const int col  = lane & 15;
    const int rb   = (lane >> 4) << 3;
#pragma unroll
    for (int i = 0; i < 2; ++i)
#pragma unroll
        for (int j = 0; j < 4; ++j) {
            const long gc = colBase + wn * 64 + j * 16 + col;
            const float bv = bias[gc];
#pragma unroll
            for (int r = 0; r < 8; ++r) {
                const long gr = rowBase + wm * 32 + i * 16 + rb + r;
                float v = acc[i][j][r] + bv;
                if constexpr (EPI == 1) v = gelu_tanh(v);
                if constexpr (EPI == 2) {
                    ((float*)Out)[(size_t)gr * N + gc] = resid[(size_t)gr * N + gc] + v;
                } else {
                    ((_Float16*)Out)[(size_t)gr * N + gc] = (_Float16)v;
                }
            }
        }
}

// ---------------------------------------------------------------------------
// Flash attention, causal, scale = 1/sqrt(H) = 1/32 (reference uses full H).
// One wave per (batch, head, 32-row query block); KV blocks of 32.
// K/V fragments loaded once per KV block and reused by both 16-row tiles.
// ---------------------------------------------------------------------------
__global__ __launch_bounds__(32) void flash_attn(const _Float16* __restrict__ Q,
                                                 const _Float16* __restrict__ Km,
                                                 const _Float16* __restrict__ V,
                                                 float* __restrict__ Out) {
    __shared__ __align__(16) _Float16 Psh[32 * 40];

    const int lane = threadIdx.x & 31;
    const int col  = lane & 15;
    const int hi   = lane >> 4;
    const int rb   = hi << 3;

    const int q0 = blockIdx.x * 32;
    const int ho = blockIdx.y * HDIM;
    const size_t rowb = (size_t)blockIdx.z * TT;   // row offset in [M,H]

    // Q fragments: two 16-row tiles x two 16x32 A-frags (K = head dim)
    v16h aq[2][2];
    {
        const int kh = hi << 4;
#pragma unroll
        for (int it = 0; it < 2; ++it)
#pragma unroll
            for (int half = 0; half < 2; ++half) {
                const _Float16* p =
                    Q + (rowb + q0 + it * 16 + col) * HH + ho + half * 32 + kh;
#pragma unroll
                for (int i = 0; i < 8; ++i) aq[it][half][i] = p[i];
#pragma unroll
                for (int i = 0; i < 8; ++i) aq[it][half][8 + i] = p[8 + i];
            }
    }

    v8f o[2][4];
#pragma unroll
    for (int it = 0; it < 2; ++it)
#pragma unroll
        for (int t = 0; t < 4; ++t) o[it][t] = zero8();
    float mrow[2][8], lrow[2][8];
#pragma unroll
    for (int it = 0; it < 2; ++it)
#pragma unroll
        for (int r = 0; r < 8; ++r) { mrow[it][r] = -1e30f; lrow[it][r] = 0.0f; }

    const float sc = 0.03125f;  // 1/sqrt(1024)

    for (int j0 = 0; j0 < q0 + 32; j0 += 32) {
        // K^T B-fragments, shared by both row tiles: [col tile][head-dim half]
        v16h bk[2][2];
#pragma unroll
        for (int nt = 0; nt < 2; ++nt)
#pragma unroll
            for (int dh = 0; dh < 2; ++dh) {
                const _Float16* p =
                    Km + (rowb + j0 + nt * 16 + col) * HH + ho + dh * 32 + (hi << 4);
#pragma unroll
                for (int i = 0; i < 16; ++i) bk[nt][dh][i] = p[i];
            }

        // S = Q @ K^T : 4 C tiles, 8 WMMAs
        v8f s[2][2];
#pragma unroll
        for (int it = 0; it < 2; ++it)
#pragma unroll
            for (int nt = 0; nt < 2; ++nt) {
                s[it][nt] = zero8();
#pragma unroll
                for (int dh = 0; dh < 2; ++dh)
                    s[it][nt] = __builtin_amdgcn_wmma_f32_16x16x32_f16(
                        false, aq[it][dh], false, bk[nt][dh], (short)0,
                        s[it][nt], false, false);
            }

        // scale + causal mask + online softmax per row tile
#pragma unroll
        for (int it = 0; it < 2; ++it) {
#pragma unroll
            for (int nt = 0; nt < 2; ++nt)
#pragma unroll
                for (int r = 0; r < 8; ++r) {
                    const int jj = j0 + nt * 16 + col;
                    const int qq = q0 + it * 16 + rb + r;
                    s[it][nt][r] = (jj <= qq) ? s[it][nt][r] * sc : -1e30f;
                }
#pragma unroll
            for (int r = 0; r < 8; ++r) {
                float rm = fmaxf(s[it][0][r], s[it][1][r]);
#pragma unroll
                for (int off = 8; off >= 1; off >>= 1)
                    rm = fmaxf(rm, __shfl_xor(rm, off, 32));
                const float mnew  = fmaxf(mrow[it][r], rm);
                const float alpha = __expf(mrow[it][r] - mnew);
                const float p0 = __expf(s[it][0][r] - mnew);
                const float p1 = __expf(s[it][1][r] - mnew);
                s[it][0][r] = p0; s[it][1][r] = p1;
                float rs = p0 + p1;
#pragma unroll
                for (int off = 8; off >= 1; off >>= 1)
                    rs += __shfl_xor(rs, off, 32);
                lrow[it][r] = lrow[it][r] * alpha + rs;
                mrow[it][r] = mnew;
#pragma unroll
                for (int t = 0; t < 4; ++t) o[it][t][r] *= alpha;
            }
            // stage P tile (16x32 f16) through LDS (C layout -> A fragment)
#pragma unroll
            for (int nt = 0; nt < 2; ++nt)
#pragma unroll
                for (int r = 0; r < 8; ++r)
                    Psh[(it * 16 + rb + r) * 40 + nt * 16 + col] =
                        (_Float16)s[it][nt][r];
        }
        __syncthreads();

        v16h ap[2];
#pragma unroll
        for (int it = 0; it < 2; ++it) ap[it] = load_frag_a(&Psh[it * 16 * 40], 40);

        // O += P @ V : V fragments shared by both row tiles
#pragma unroll
        for (int t = 0; t < 4; ++t) {
            v16h bv;
            const _Float16* p = V + (rowb + j0 + (hi << 4)) * HH + ho + t * 16 + col;
#pragma unroll
            for (int i = 0; i < 16; ++i) bv[i] = p[(size_t)i * HH];
#pragma unroll
            for (int it = 0; it < 2; ++it)
                o[it][t] = __builtin_amdgcn_wmma_f32_16x16x32_f16(
                    false, ap[it], false, bv, (short)0, o[it][t], false, false);
        }
        __syncthreads();
    }

    // normalize and write f32 attention output
#pragma unroll
    for (int it = 0; it < 2; ++it)
#pragma unroll
        for (int t = 0; t < 4; ++t)
#pragma unroll
            for (int r = 0; r < 8; ++r)
                Out[(rowb + q0 + it * 16 + rb + r) * HH + ho + t * 16 + col] =
                    o[it][t][r] / lrow[it][r];
}

// ---------------------------------------------------------------------------
// Host-side orchestration
// ---------------------------------------------------------------------------
extern "C" void kernel_launch(void* const* d_in, const int* in_sizes, int n_in,
                              void* d_out, int out_size, void* d_ws, size_t ws_size,
                              hipStream_t stream) {
    (void)in_sizes; (void)n_in; (void)out_size; (void)ws_size;

    const float* x  = (const float*)d_in[0];
    const float* Wq = (const float*)d_in[1];
    const float* bq = (const float*)d_in[2];
    const float* Wk = (const float*)d_in[3];
    const float* bk = (const float*)d_in[4];
    const float* Wv = (const float*)d_in[5];
    const float* bv = (const float*)d_in[6];
    const float* W1 = (const float*)d_in[7];
    const float* b1 = (const float*)d_in[8];
    const float* W2 = (const float*)d_in[9];
    const float* b2 = (const float*)d_in[10];

    char* ws = (char*)d_ws;
    size_t off = 0;
    auto take = [&](size_t bytes) {
        char* p = ws + off;
        off = (off + bytes + 255) & ~(size_t)255;
        return p;
    };

    const size_t nXH = (size_t)MM * HH;       // 8,388,608
    const size_t nW  = (size_t)HH * HH;       // 1,048,576
    const size_t nW1 = (size_t)HH * FFDIM;    // 4,194,304
    const size_t nH1 = (size_t)MM * FFDIM;    // 33,554,432

    _Float16* xh  = (_Float16*)take(nXH * 2);
    _Float16* wqh = (_Float16*)take(nW * 2);
    _Float16* wkh = (_Float16*)take(nW * 2);
    _Float16* wvh = (_Float16*)take(nW * 2);
    _Float16* w1h = (_Float16*)take(nW1 * 2);
    _Float16* w2h = (_Float16*)take(nW1 * 2);
    _Float16* qh  = (_Float16*)take(nXH * 2);
    _Float16* kh  = (_Float16*)take(nXH * 2);
    _Float16* vh  = (_Float16*)take(nXH * 2);
    float*    axr = (float*)take(nXH * 4);    // attn out, then xr in place
    _Float16* xrh = (_Float16*)take(nXH * 2);
    _Float16* h1h = (_Float16*)take(nH1 * 2);

    const dim3 blk(256);
    cvt_f32_to_f16<<<dim3(2048), blk, 0, stream>>>(x,  xh,  nXH);
    cvt_f32_to_f16<<<dim3(512),  blk, 0, stream>>>(Wq, wqh, nW);
    cvt_f32_to_f16<<<dim3(512),  blk, 0, stream>>>(Wk, wkh, nW);
    cvt_f32_to_f16<<<dim3(512),  blk, 0, stream>>>(Wv, wvh, nW);
    cvt_f32_to_f16<<<dim3(1024), blk, 0, stream>>>(W1, w1h, nW1);
    cvt_f32_to_f16<<<dim3(1024), blk, 0, stream>>>(W2, w2h, nW1);

    const dim3 gQKV(HH / 128, MM / 128);
    gemm_wmma<0><<<gQKV, blk, 0, stream>>>(xh, wqh, bq, qh, nullptr, MM, HH, HH);
    gemm_wmma<0><<<gQKV, blk, 0, stream>>>(xh, wkh, bk, kh, nullptr, MM, HH, HH);
    gemm_wmma<0><<<gQKV, blk, 0, stream>>>(xh, wvh, bv, vh, nullptr, MM, HH, HH);

    flash_attn<<<dim3(TT / 32, NHEAD, BB), dim3(32), 0, stream>>>(qh, kh, vh, axr);

    add_residual<<<dim3(2048), blk, 0, stream>>>(x, axr, xrh, nXH);

    gemm_wmma<1><<<dim3(FFDIM / 128, MM / 128), blk, 0, stream>>>(
        xrh, w1h, b1, h1h, nullptr, MM, FFDIM, HH);
    gemm_wmma<2><<<dim3(HH / 128, MM / 128), blk, 0, stream>>>(
        h1h, w2h, b2, d_out, axr, MM, HH, FFDIM);
}
